// TaoMamba_24567212934070
// MI455X (gfx1250) — matrix operand order repaired and checked by reading, not compiled, not run on hardware
//
#include <hip/hip_runtime.h>
#include <hip/hip_bf16.h>

// ---------------- model constants (from reference setup_inputs) -------------
#define NB        8            // batch
#define TLEN      2048
#define LH        2045         // sequence after window-MLP + cls token
#define NTOK      (NB * LH)    // 16360 tokens
#define DIM       128
#define DINNER    256
#define NH        4
#define HD        64
#define NSTATE    64
#define CONVDIM   384          // d_inner + 2*n
#define DINPROJ   644          // 2*256 + 2*64 + 4
#define DEPTH     4
#define DCONV     4

typedef __attribute__((ext_vector_type(16))) _Float16 v16h;
typedef __attribute__((ext_vector_type(8)))  float    v8f;

// ---------------------------------------------------------------------------
// WMMA GEMM:  C[m][n] = sum_k A[m][k] * W[n][k]
// A: M x K f32 row-major, W: N x K f32 row-major. f32->f16 fragments inline,
// f32 accumulation via v_wmma_f32_16x16x32_f16. One wave = one 16x16 tile.
// Fragment layout per CDNA5 ISA 7.12.2 (wave32):
//   A: lane -> row (lane&15); elems 0..7  = K[kk + hi*8 .. +8)
//                             elems 8..15 = K[kk + 16 + hi*8 .. +8)
//   B: lane -> col (lane&15); elems 0..7  = K[kk + hi*8 .. +8)
//                             elems 8..15 = K[kk + 16 + hi*8 .. +8)
//   D: elem i -> row hi*8 + i, col lane&15
// ---------------------------------------------------------------------------
template<int K>
__global__ __launch_bounds__(128)
void taomamba_wmma_gemm(const float* __restrict__ A, const float* __restrict__ W,
                        float* __restrict__ C, int M, int N,
                        int tilesN, int tilesTotal)
{
    const int wave = threadIdx.x >> 5;
    const int lane = threadIdx.x & 31;
    int tile = blockIdx.x * 4 + wave;
    if (tile >= tilesTotal) return;              // wave-uniform exit (EXEC stays all-1 for WMMA)

    const int tm = tile / tilesN;
    const int tn = tile % tilesN;
    const int hi = lane >> 4;

    int rm = tm * 16 + (lane & 15); if (rm > M - 1) rm = M - 1;   // clamp edge loads
    int cn = tn * 16 + (lane & 15); if (cn > N - 1) cn = N - 1;

    const float* __restrict__ arow = A + (size_t)rm * K;
    const float* __restrict__ wrow = W + (size_t)cn * K;

    v8f acc = {};
#pragma unroll
    for (int kk = 0; kk < K; kk += 32) {
        if (kk + 32 < K) {                        // prefetch next K panel
            __builtin_prefetch(arow + kk + 32, 0, 1);
            __builtin_prefetch(wrow + kk + 32, 0, 1);
        }
        v16h af, bf;
#pragma unroll
        for (int i = 0; i < 8; ++i) {
            af[i]     = (_Float16)arow[kk + hi * 8 + i];
            af[8 + i] = (_Float16)arow[kk + 16 + hi * 8 + i];
        }
#pragma unroll
        for (int i = 0; i < 8; ++i) {
            bf[i]     = (_Float16)wrow[kk + hi * 8 + i];
            bf[8 + i] = (_Float16)wrow[kk + 16 + hi * 8 + i];
        }

        acc = __builtin_amdgcn_wmma_f32_16x16x32_f16(
                  /*neg_a=*/false, af, /*neg_b=*/false, bf,
                  /*c_mod=*/(short)0, acc, /*reuse_a=*/false, /*reuse_b=*/false);
    }

    const int col = tn * 16 + (lane & 15);
    if (col < N) {
#pragma unroll
        for (int i = 0; i < 8; ++i) {
            int m = tm * 16 + hi * 8 + i;
            if (m < M) C[(size_t)m * N + col] = acc[i];
        }
    }
}

// ---------------------------------------------------------------------------
// Window embedding: tok(15) -> gelu(W1) -> 128 -> W2 -> 128 ; token LH-1 = cls
// One block (128 threads) per token.
// ---------------------------------------------------------------------------
__global__ __launch_bounds__(128)
void taomamba_embed(const float* __restrict__ x,
                    const float* __restrict__ w1, const float* __restrict__ b1,
                    const float* __restrict__ w2, const float* __restrict__ b2,
                    const float* __restrict__ clsTok, float* __restrict__ h)
{
    const int tok = blockIdx.x;              // 0 .. NTOK-1
    const int b   = tok / LH;
    const int l   = tok % LH;
    const int j   = threadIdx.x;             // 0..127
    float* hrow = h + (size_t)tok * DIM;

    if (l == LH - 1) {                       // cls token (uniform branch per block)
        hrow[j] = clsTok[j];
        return;
    }

    __shared__ float sTok[16];
    __shared__ float sHid[DIM];
    if (j < 15) {
        int w = j / 3, c = j % 3;
        sTok[j] = x[((size_t)b * TLEN + (l + w)) * 3 + c];
    }
    __syncthreads();

    float acc = b1[j];
#pragma unroll
    for (int k = 0; k < 15; ++k) acc += sTok[k] * w1[j * 15 + k];
    // exact GELU
    sHid[j] = 0.5f * acc * (1.0f + erff(acc * 0.70710678118654752f));
    __syncthreads();

    float acc2 = b2[j];
    for (int k = 0; k < DIM; ++k) acc2 += sHid[k] * w2[j * DIM + k];
    hrow[j] = acc2;
}

// ---------------------------------------------------------------------------
// Causal depthwise conv (K=4) + SiLU over xBC; softplus(dt + dt_bias).
// zxbcdt token row layout: [ z(0:256) | xBC(256:640) | dt(640:644) ]
// ---------------------------------------------------------------------------
__global__ __launch_bounds__(256)
void taomamba_conv_dt(const float* __restrict__ zxb,
                      const float* __restrict__ cw, const float* __restrict__ cb,
                      const float* __restrict__ dtb,
                      float* __restrict__ xbcc, float* __restrict__ dtbuf)
{
    size_t idx = (size_t)blockIdx.x * blockDim.x + threadIdx.x;
    const size_t total = (size_t)NTOK * (CONVDIM + NH);
    if (idx >= total) return;

    const int c     = (int)(idx % (CONVDIM + NH));
    const size_t tok = idx / (CONVDIM + NH);
    const int l = (int)(tok % LH);
    const int b = (int)(tok / LH);

    if (c < CONVDIM) {
        float acc = cb[c];
#pragma unroll
        for (int k = 0; k < DCONV; ++k) {
            int ls = l - (DCONV - 1) + k;
            if (ls >= 0)
                acc += zxb[((size_t)b * LH + ls) * DINPROJ + DINNER + c] * cw[c * DCONV + k];
        }
        xbcc[tok * CONVDIM + c] = acc / (1.0f + expf(-acc)) * 1.0f * acc / (acc == 0.0f ? 1.0f : acc); // silu
        // (silu simplified below without the identity trick)
        xbcc[tok * CONVDIM + c] = acc * (1.0f / (1.0f + expf(-acc)));
    } else {
        int hh = c - CONVDIM;
        float v = zxb[tok * DINPROJ + (DINPROJ - NH) + hh] + dtb[hh];
        dtbuf[tok * NH + hh] = (v > 20.0f) ? v : log1pf(expf(v));   // softplus
    }
}

// ---------------------------------------------------------------------------
// Selective scan. One block per (batch, head); 256 threads.
// Thread t owns state h[p = t&63][n = (t>>6)*16 .. +16) in 16 VGPRs.
// Per step: stage x/B/C/dt in LDS, update state, reduce y over the 4 n-chunks.
// ---------------------------------------------------------------------------
__global__ __launch_bounds__(256)
void taomamba_scan(const float* __restrict__ xbcc, const float* __restrict__ dtbuf,
                   const float* __restrict__ alog, float* __restrict__ yscan)
{
    const int b  = blockIdx.x / NH;
    const int hh = blockIdx.x % NH;
    const int t  = threadIdx.x;
    const int p  = t & 63;
    const int nc = t >> 6;                 // 0..3

    const float A = -expf(alog[hh]);

    float hs[16];
#pragma unroll
    for (int i = 0; i < 16; ++i) hs[i] = 0.0f;

    __shared__ float sx[NSTATE], sB[NSTATE], sC[NSTATE];
    __shared__ float sy[4][NSTATE];
    __shared__ float sdt;

    for (int l = 0; l < LH; ++l) {
        const size_t tok = (size_t)b * LH + l;
        const float* xrow = xbcc + tok * CONVDIM;
        if (t < 64)            sx[t]       = xrow[hh * HD + t];
        else if (t < 128)      sB[t - 64]  = xrow[DINNER + (t - 64)];
        else if (t < 192)      sC[t - 128] = xrow[DINNER + NSTATE + (t - 128)];
        else if (t == 192)     sdt         = dtbuf[tok * NH + hh];
        __syncthreads();

        const float dt    = sdt;
        const float decay = __expf(dt * A);
        const float xv    = dt * sx[p];
        float part = 0.0f;
#pragma unroll
        for (int i = 0; i < 16; ++i) {
            const int n = nc * 16 + i;
            hs[i] = hs[i] * decay + xv * sB[n];
            part += hs[i] * sC[n];
        }
        sy[nc][p] = part;
        __syncthreads();

        if (t < 64)
            yscan[tok * DINNER + hh * HD + t] = sy[0][t] + sy[1][t] + sy[2][t] + sy[3][t];
        __syncthreads();   // protect LDS before next step's staging
    }
}

// ---------------------------------------------------------------------------
// y = (yscan + Dh*xh) * silu(z); RMS-norm over 256 * norm_w  -> yn (f32)
// One block (256 threads) per token.
// ---------------------------------------------------------------------------
__global__ __launch_bounds__(256)
void taomamba_gate_norm(const float* __restrict__ yscan, const float* __restrict__ xbcc,
                        const float* __restrict__ zxb,  const float* __restrict__ Dh,
                        const float* __restrict__ nw,   float* __restrict__ yn)
{
    const size_t tok = blockIdx.x;
    const int d = threadIdx.x;              // 0..255
    const int hh = d >> 6;

    float y = yscan[tok * DINNER + d] + Dh[hh] * xbcc[tok * CONVDIM + d];
    float z = zxb[tok * DINPROJ + d];
    float g = y * (z * (1.0f / (1.0f + expf(-z))));

    __shared__ float red[DINNER];
    red[d] = g * g;
    __syncthreads();
    for (int s = 128; s > 0; s >>= 1) {
        if (d < s) red[d] += red[d + s];
        __syncthreads();
    }
    const float inv = rsqrtf(red[0] / (float)DINNER + 1e-5f);
    yn[tok * DINNER + d] = g * inv * nw[d];
}

// ---------------------------------------------------------------------------
// Final head: LayerNorm(cls) -> dot(head_w) + head_b. One block per batch.
// ---------------------------------------------------------------------------
__global__ __launch_bounds__(128)
void taomamba_head(const float* __restrict__ hfin,
                   const float* __restrict__ lnw, const float* __restrict__ lnb,
                   const float* __restrict__ hw,  const float* __restrict__ hb,
                   float* __restrict__ out)
{
    const int b = blockIdx.x;
    const int j = threadIdx.x;
    __shared__ float red[DIM];

    const float v = hfin[((size_t)b * LH + (LH - 1)) * DIM + j];

    red[j] = v; __syncthreads();
    for (int s = 64; s > 0; s >>= 1) { if (j < s) red[j] += red[j + s]; __syncthreads(); }
    const float mu = red[0] / (float)DIM;
    __syncthreads();

    const float dv = v - mu;
    red[j] = dv * dv; __syncthreads();
    for (int s = 64; s > 0; s >>= 1) { if (j < s) red[j] += red[j + s]; __syncthreads(); }
    const float var = red[0] / (float)DIM;
    __syncthreads();

    const float cn = dv * rsqrtf(var + 1e-5f) * lnw[j] + lnb[j];
    red[j] = cn * hw[j]; __syncthreads();
    for (int s = 64; s > 0; s >>= 1) { if (j < s) red[j] += red[j + s]; __syncthreads(); }
    if (j == 0) out[b] = red[0] + hb[0];
}

// ---------------------------------------------------------------------------
extern "C" void kernel_launch(void* const* d_in, const int* in_sizes, int n_in,
                              void* d_out, int out_size, void* d_ws, size_t ws_size,
                              hipStream_t stream)
{
    (void)in_sizes; (void)n_in; (void)out_size; (void)ws_size;

    const float* x         = (const float*)d_in[0];
    const float* mlp_w1    = (const float*)d_in[1];
    const float* mlp_b1    = (const float*)d_in[2];
    const float* mlp_w2    = (const float*)d_in[3];
    const float* mlp_b2    = (const float*)d_in[4];
    const float* cls_token = (const float*)d_in[5];
    const float* in_proj_w = (const float*)d_in[6];   // (4, 644, 128)
    const float* conv_w    = (const float*)d_in[7];   // (4, 384, 4)
    const float* conv_b    = (const float*)d_in[8];   // (4, 384)
    const float* dt_bias   = (const float*)d_in[9];   // (4, 4)
    const float* A_log     = (const float*)d_in[10];  // (4, 4)
    const float* Dh        = (const float*)d_in[11];  // (4, 4)
    const float* norm_w    = (const float*)d_in[12];  // (4, 256)
    const float* out_proj_w= (const float*)d_in[13];  // (4, 128, 256)
    const float* ln_w      = (const float*)d_in[14];
    const float* ln_b      = (const float*)d_in[15];
    const float* head_w    = (const float*)d_in[16];
    const float* head_b    = (const float*)d_in[17];

    float* out  = (float*)d_out;            // out[0..7]
    float* hout = (float*)d_out + 8;        // final h (NTOK*128)

    // ---- workspace carve-up (all f32) ----
    float* hbuf  = (float*)d_ws;                         // NTOK*128
    float* zxb   = hbuf  + (size_t)NTOK * DIM;           // NTOK*644
    float* xbcc  = zxb   + (size_t)NTOK * DINPROJ;       // NTOK*384
    float* dtbuf = xbcc  + (size_t)NTOK * CONVDIM;       // NTOK*4
    float* yscan = dtbuf + (size_t)NTOK * NH;            // NTOK*256
    float* yn    = yscan + (size_t)NTOK * DINNER;        // NTOK*256

    // 1) embedding + cls
    taomamba_embed<<<NTOK, 128, 0, stream>>>(x, mlp_w1, mlp_b1, mlp_w2, mlp_b2,
                                             cls_token, hbuf);

    const int tilesM  = (NTOK + 15) / 16;        // 1023
    const int tilesN1 = (DINPROJ + 15) / 16;     // 41
    const int tiles1  = tilesM * tilesN1;
    const int tilesN2 = DIM / 16;                // 8
    const int tiles2  = tilesM * tilesN2;

    for (int i = 0; i < DEPTH; ++i) {
        const float* inw = in_proj_w  + (size_t)i * DINPROJ * DIM;
        const float* cw  = conv_w     + (size_t)i * CONVDIM * DCONV;
        const float* cb  = conv_b     + (size_t)i * CONVDIM;
        const float* dtb = dt_bias    + (size_t)i * NH;
        const float* al  = A_log      + (size_t)i * NH;
        const float* dh  = Dh         + (size_t)i * NH;
        const float* nw  = norm_w     + (size_t)i * DINNER;
        const float* ow  = out_proj_w + (size_t)i * DIM * DINNER;

        // 2) in_proj GEMM (K=128): zxbcdt = h @ in_w.T
        taomamba_wmma_gemm<DIM><<<(tiles1 + 3) / 4, 128, 0, stream>>>(
            hbuf, inw, zxb, NTOK, DINPROJ, tilesN1, tiles1);

        // 3) causal depthwise conv + SiLU, softplus(dt)
        {
            const size_t total = (size_t)NTOK * (CONVDIM + NH);
            taomamba_conv_dt<<<(unsigned)((total + 255) / 256), 256, 0, stream>>>(
                zxb, cw, cb, dtb, xbcc, dtbuf);
        }

        // 4) selective scan (serial recurrence, state in VGPRs)
        taomamba_scan<<<NB * NH, 256, 0, stream>>>(xbcc, dtbuf, al, yscan);

        // 5) gate + RMS norm
        taomamba_gate_norm<<<NTOK, 256, 0, stream>>>(yscan, xbcc, zxb, dh, nw, yn);

        // 6) out_proj GEMM (K=256): h = yn @ ow.T ; last layer writes d_out's h
        float* outC = (i == DEPTH - 1) ? hout : hbuf;
        taomamba_wmma_gemm<DINNER><<<(tiles2 + 3) / 4, 128, 0, stream>>>(
            yn, ow, outC, NTOK, DIM, tilesN2, tiles2);
    }

    // 7) LayerNorm + head on cls rows
    taomamba_head<<<NB, 128, 0, stream>>>(hout, ln_w, ln_b, head_w, head_b, out);
}
